// SelfAttention_89635967467864
// MI455X (gfx1250) — matrix operand order repaired and checked by reading, not compiled, not run on hardware
//
#include <hip/hip_runtime.h>
#include <hip/hip_bf16.h>

typedef __attribute__((ext_vector_type(16))) _Float16 v16h;
typedef __attribute__((ext_vector_type(8)))  _Float16 v8h;
typedef __attribute__((ext_vector_type(8)))  float    v8f;

namespace {
constexpr int B = 2, N = 4096, C = 256, H = 4, DH = 64;
constexpr int BH = B * H;                 // 8
constexpr int NSL1 = 8;                   // pass-1 i-reduction slices
constexpr int NSL2 = 2;                   // pass-2 j-contraction slices
constexpr float SCALE = 0.125f;           // 1/sqrt(DH)
constexpr float EPS = 1e-6f;

constexpr size_t SZ_H16 = (size_t)B * N * C * sizeof(_Float16);      // 4 MiB
constexpr size_t SZ_QKV = (size_t)B * H * N * DH * sizeof(_Float16); // 4 MiB
constexpr size_t SZ_W   = (size_t)H * C * DH * sizeof(_Float16);     // 128 KiB
constexpr size_t SZ_WO  = (size_t)C * C * sizeof(_Float16);          // 128 KiB
constexpr size_t SZ_ST  = (size_t)BH * N * sizeof(float);            // 128 KiB
constexpr size_t SZ_P1  = (size_t)NSL1 * BH * N * sizeof(float);     // 1 MiB
constexpr size_t SZ_OP  = (size_t)NSL2 * BH * N * DH * sizeof(float);// 16 MiB

constexpr size_t OFF_H   = 0;
constexpr size_t OFF_Q   = OFF_H   + SZ_H16;
constexpr size_t OFF_K   = OFF_Q   + SZ_QKV;
constexpr size_t OFF_VT  = OFF_K   + SZ_QKV;
constexpr size_t OFF_O   = OFF_VT  + SZ_QKV;
constexpr size_t OFF_WQ  = OFF_O   + SZ_H16;
constexpr size_t OFF_WK  = OFF_WQ  + SZ_W;
constexpr size_t OFF_WV  = OFF_WK  + SZ_W;
constexpr size_t OFF_WO  = OFF_WV  + SZ_W;
constexpr size_t OFF_CM  = OFF_WO  + SZ_WO;
constexpr size_t OFF_CS  = OFF_CM  + SZ_ST;
constexpr size_t OFF_P1M = OFF_CS  + SZ_ST;
constexpr size_t OFF_P1S = OFF_P1M + SZ_P1;
constexpr size_t OFF_OP  = OFF_P1S + SZ_P1;
}

union AFrag { v16h v; v8h h[2]; };

__device__ __forceinline__ v8f wmma16(v16h a, v16h b, v8f c) {
  // D = A(16x32 f16) * B(32x16 f16) + C(16x16 f32)
  return __builtin_amdgcn_wmma_f32_16x16x32_f16(false, a, false, b, (short)0, c,
                                                false, false);
}

// ---- weight pre-transpose: W[H][C][DH] f32 -> Wt[H][DH][C] f16 (K-contiguous columns)
__global__ void __launch_bounds__(256) k_trans_hcd(const float* __restrict__ W,
                                                   _Float16* __restrict__ Wt) {
  const int idx = blockIdx.x * 256 + threadIdx.x;            // < H*C*DH
  const int hd  = idx / (C * DH);
  const int rem = idx - hd * C * DH;
  const int c   = rem / DH;
  const int d   = rem - c * DH;
  Wt[((size_t)hd * DH + d) * C + c] = (_Float16)W[idx];
}

// ---- Wo[C][C] (ci,co) f32 -> Wt[co][ci] f16
__global__ void __launch_bounds__(256) k_trans_cc(const float* __restrict__ W,
                                                  _Float16* __restrict__ Wt) {
  const int idx = blockIdx.x * 256 + threadIdx.x;            // < C*C
  const int ci = idx / C;
  const int co = idx - ci * C;
  Wt[(size_t)co * C + ci] = (_Float16)W[idx];
}

// ---- fused (x + pos) -> LayerNorm -> f16, one row (C=256) per 256-thread block
__global__ void __launch_bounds__(256) k_ln(const float* __restrict__ x,
                                            const float* __restrict__ pe,
                                            const float* __restrict__ g,
                                            const float* __restrict__ be,
                                            _Float16* __restrict__ h16) {
  __shared__ float sm[8];
  const int row = blockIdx.x;
  const int c = threadIdx.x;
  const float v = x[(size_t)row * C + c] + pe[(size_t)row * C + c];
  float s = v;
  #pragma unroll
  for (int o = 16; o > 0; o >>= 1) s += __shfl_xor(s, o, 32);
  if ((c & 31) == 0) sm[c >> 5] = s;
  __syncthreads();
  float tot = 0.f;
  #pragma unroll
  for (int i = 0; i < 8; ++i) tot += sm[i];
  const float mean = tot * (1.0f / C);
  const float d = v - mean;
  __syncthreads();
  s = d * d;
  #pragma unroll
  for (int o = 16; o > 0; o >>= 1) s += __shfl_xor(s, o, 32);
  if ((c & 31) == 0) sm[c >> 5] = s;
  __syncthreads();
  tot = 0.f;
  #pragma unroll
  for (int i = 0; i < 8; ++i) tot += sm[i];
  const float var = tot * (1.0f / C);
  const float rs = rsqrtf(var + EPS);
  h16[(size_t)row * C + c] = (_Float16)(d * rs * g[c] + be[c]);
}

// ---- QKV projection: one wave per 16-row tile per (b,h). Q,K row-major; V transposed [d][n].
__global__ void __launch_bounds__(32) k_qkv(const _Float16* __restrict__ h16,
                                            const _Float16* __restrict__ wqt,
                                            const _Float16* __restrict__ wkt,
                                            const _Float16* __restrict__ wvt,
                                            const float* __restrict__ bq,
                                            const float* __restrict__ bk,
                                            const float* __restrict__ bv,
                                            _Float16* __restrict__ q16,
                                            _Float16* __restrict__ k16,
                                            _Float16* __restrict__ vt16) {
  const int lane = threadIdx.x;
  const int it = blockIdx.x;     // N/16 tiles
  const int bh = blockIdx.y;     // B*H
  const int b  = bh >> 2;
  const int hd = bh & 3;
  const int lhi = lane >> 4, llo = lane & 15;

  v8f aq[4] = {}, ak[4] = {}, av[4] = {};
  const _Float16* hrow = h16 + ((size_t)b * N + (it * 16 + llo)) * C;

  for (int kc = 0; kc < C; kc += 32) {
    AFrag a;
    a.h[0] = *(const v8h*)(hrow + kc + (lhi << 4));
    a.h[1] = *(const v8h*)(hrow + kc + 8 + (lhi << 4));
    #pragma unroll
    for (int db = 0; db < 4; ++db) {
      const int d = db * 16 + llo;
      const size_t wrow = ((size_t)hd * DH + d) * C + kc + (lhi << 4);
      const v16h bqf = *(const v16h*)(wqt + wrow);
      const v16h bkf = *(const v16h*)(wkt + wrow);
      const v16h bvf = *(const v16h*)(wvt + wrow);
      aq[db] = wmma16(a.v, bqf, aq[db]);
      ak[db] = wmma16(a.v, bkf, ak[db]);
      av[db] = wmma16(a.v, bvf, av[db]);
    }
  }
  #pragma unroll
  for (int db = 0; db < 4; ++db) {
    const int d = db * 16 + llo;
    const float biq = bq[hd * DH + d];
    const float bik = bk[hd * DH + d];
    const float biv = bv[hd * DH + d];
    #pragma unroll
    for (int r = 0; r < 8; ++r) {
      const int n = it * 16 + r + (lhi << 3);
      q16[((size_t)bh * N + n) * DH + d] = (_Float16)(aq[db][r] + biq);
      k16[((size_t)bh * N + n) * DH + d] = (_Float16)(ak[db][r] + bik);
      vt16[((size_t)bh * DH + d) * N + n] = (_Float16)(av[db][r] + biv);
    }
  }
}

// ---- pass 1: per-column (query-axis) online max + sum(exp) of S = QK^T/sqrt(dh).
// 32 columns per wave (two loop-invariant B-fragment pairs -> 4 WMMA per A-load);
// i-reduction sliced NSL1 ways for occupancy; partials merged by k_combine.
__global__ void __launch_bounds__(32) k_pass1(const _Float16* __restrict__ q16,
                                              const _Float16* __restrict__ k16,
                                              float* __restrict__ pmax,
                                              float* __restrict__ psum) {
  const int lane = threadIdx.x;
  const int jt = blockIdx.x;     // N/32 column groups
  const int bh = blockIdx.y;
  const int sl = blockIdx.z;     // i slice
  const int lhi = lane >> 4, llo = lane & 15;
  const int j0 = jt * 32 + llo;
  const int j1 = j0 + 16;

  // loop-invariant B fragments: columns j of K^T = rows j of K (d-contiguous)
  const _Float16* kr0 = k16 + ((size_t)bh * N + j0) * DH + (lhi << 4);
  const _Float16* kr1 = k16 + ((size_t)bh * N + j1) * DH + (lhi << 4);
  const v16h b00 = *(const v16h*)(kr0);
  const v16h b01 = *(const v16h*)(kr0 + 32);
  const v16h b10 = *(const v16h*)(kr1);
  const v16h b11 = *(const v16h*)(kr1 + 32);

  float m0 = -3.0e38f, l0 = 0.0f, m1 = -3.0e38f, l1 = 0.0f;
  const int ibeg = sl * (N / NSL1);
  const int iend = ibeg + (N / NSL1);
  for (int i0 = ibeg; i0 < iend; i0 += 16) {
    const _Float16* qrow = q16 + ((size_t)bh * N + (i0 + llo)) * DH;
    AFrag a0, a1;
    a0.h[0] = *(const v8h*)(qrow + (lhi << 4));
    a0.h[1] = *(const v8h*)(qrow + 8 + (lhi << 4));
    a1.h[0] = *(const v8h*)(qrow + 32 + (lhi << 4));
    a1.h[1] = *(const v8h*)(qrow + 40 + (lhi << 4));
    v8f s0 = {}, s1 = {};
    s0 = wmma16(a0.v, b00, s0);
    s0 = wmma16(a1.v, b01, s0);
    s1 = wmma16(a0.v, b10, s1);
    s1 = wmma16(a1.v, b11, s1);

    float sc[8], tm;
    // tile 0 online update
    tm = -3.0e38f;
    #pragma unroll
    for (int r = 0; r < 8; ++r) { sc[r] = s0[r] * SCALE; tm = fmaxf(tm, sc[r]); }
    {
      const float nm = fmaxf(m0, tm);
      float acc = 0.0f;
      #pragma unroll
      for (int r = 0; r < 8; ++r) acc += __expf(sc[r] - nm);
      l0 = l0 * __expf(m0 - nm) + acc;
      m0 = nm;
    }
    // tile 1 online update
    tm = -3.0e38f;
    #pragma unroll
    for (int r = 0; r < 8; ++r) { sc[r] = s1[r] * SCALE; tm = fmaxf(tm, sc[r]); }
    {
      const float nm = fmaxf(m1, tm);
      float acc = 0.0f;
      #pragma unroll
      for (int r = 0; r < 8; ++r) acc += __expf(sc[r] - nm);
      l1 = l1 * __expf(m1 - nm) + acc;
      m1 = nm;
    }
  }
  // lanes j and j+16 saw disjoint row halves of each tile; merge them
  {
    const float mo = __shfl_xor(m0, 16, 32), lo = __shfl_xor(l0, 16, 32);
    const float nm = fmaxf(m0, mo);
    const float L = l0 * __expf(m0 - nm) + lo * __expf(mo - nm);
    if (lane < 16) {
      pmax[((size_t)sl * BH + bh) * N + j0] = nm;
      psum[((size_t)sl * BH + bh) * N + j0] = L;
    }
  }
  {
    const float mo = __shfl_xor(m1, 16, 32), lo = __shfl_xor(l1, 16, 32);
    const float nm = fmaxf(m1, mo);
    const float L = l1 * __expf(m1 - nm) + lo * __expf(mo - nm);
    if (lane < 16) {
      pmax[((size_t)sl * BH + bh) * N + j1] = nm;
      psum[((size_t)sl * BH + bh) * N + j1] = L;
    }
  }
}

// ---- merge pass-1 slice partials into final colmax/colsum
__global__ void __launch_bounds__(256) k_combine(const float* __restrict__ pmax,
                                                 const float* __restrict__ psum,
                                                 float* __restrict__ cmax,
                                                 float* __restrict__ csum) {
  const size_t idx = (size_t)blockIdx.x * 256 + threadIdx.x;  // < BH*N
  constexpr size_t STRIDE = (size_t)BH * N;
  float m = -3.0e38f;
  #pragma unroll
  for (int s = 0; s < NSL1; ++s) m = fmaxf(m, pmax[s * STRIDE + idx]);
  float L = 0.0f;
  #pragma unroll
  for (int s = 0; s < NSL1; ++s)
    L += psum[s * STRIDE + idx] * __expf(pmax[s * STRIDE + idx] - m);
  cmax[idx] = m;
  csum[idx] = L;
}

// ---- scale V^T rows by 1/colsum[j]
__global__ void __launch_bounds__(256) k_scalev(_Float16* __restrict__ vt16,
                                                const float* __restrict__ csum) {
  const size_t idx = (size_t)blockIdx.x * 256 + threadIdx.x;  // < B*H*DH*N
  const int j  = (int)(idx & (size_t)(N - 1));
  const int bh = (int)(idx / ((size_t)DH * N));
  const float rs = 1.0f / csum[(size_t)bh * N + j];
  vt16[idx] = (_Float16)((float)vt16[idx] * rs);
}

// ---- pass 2: O = exp(S - colmax) @ (V/colsum); j contraction sliced NSL2 ways
// into f32 partial-O buffers; P re-layout to A-fragment via 1 KB LDS transpose.
__global__ void __launch_bounds__(32) k_pass2(const _Float16* __restrict__ q16,
                                              const _Float16* __restrict__ k16,
                                              const _Float16* __restrict__ vt16,
                                              const float* __restrict__ cmax,
                                              float* __restrict__ opart) {
  __shared__ alignas(32) _Float16 pt[16 * 32];
  const int lane = threadIdx.x;
  const int it = blockIdx.x;     // N/16 row tiles
  const int bh = blockIdx.y;
  const int sl = blockIdx.z;     // j slice
  const int lhi = lane >> 4, llo = lane & 15;

  // loop-invariant A fragments (Q rows of this tile)
  const _Float16* qrow = q16 + ((size_t)bh * N + (it * 16 + llo)) * DH;
  AFrag a0, a1;
  a0.h[0] = *(const v8h*)(qrow + (lhi << 4));
  a0.h[1] = *(const v8h*)(qrow + 8 + (lhi << 4));
  a1.h[0] = *(const v8h*)(qrow + 32 + (lhi << 4));
  a1.h[1] = *(const v8h*)(qrow + 40 + (lhi << 4));

  v8f o[4] = {};
  const int jbeg = sl * (N / NSL2);
  const int jend = jbeg + (N / NSL2);
  for (int jj = jbeg; jj < jend; jj += 32) {
    #pragma unroll
    for (int jh = 0; jh < 2; ++jh) {
      const int j = jj + jh * 16 + llo;
      const _Float16* krow = k16 + ((size_t)bh * N + j) * DH + (lhi << 4);
      v8f s = {};
      s = wmma16(a0.v, *(const v16h*)(krow), s);
      s = wmma16(a1.v, *(const v16h*)(krow + 32), s);
      const float cm = cmax[(size_t)bh * N + j];
      #pragma unroll
      for (int r = 0; r < 8; ++r)
        pt[(r + (lhi << 3)) * 32 + jh * 16 + llo] =
            (_Float16)__expf(s[r] * SCALE - cm);
    }
    // same-wave DS ops are in-order; wait + compiler barrier before re-reading
    asm volatile("s_wait_dscnt 0" ::: "memory");
    AFrag pa;
    pa.h[0] = *(const v8h*)(&pt[llo * 32 + (lhi << 4)]);
    pa.h[1] = *(const v8h*)(&pt[llo * 32 + 8 + (lhi << 4)]);
    #pragma unroll
    for (int db = 0; db < 4; ++db) {
      const int d = db * 16 + llo;
      const v16h bvf =
          *(const v16h*)(vt16 + ((size_t)bh * DH + d) * N + jj + (lhi << 4));
      o[db] = wmma16(pa.v, bvf, o[db]);
    }
    asm volatile("" ::: "memory");
  }
  float* op = opart + (size_t)sl * BH * N * DH;
  #pragma unroll
  for (int db = 0; db < 4; ++db) {
    #pragma unroll
    for (int r = 0; r < 8; ++r) {
      const int n = it * 16 + r + (lhi << 3);
      op[((size_t)bh * N + n) * DH + db * 16 + llo] = o[db][r];
    }
  }
}

// ---- merge pass-2 slice partials -> head-concatenated f16 O[b][n][c]
__global__ void __launch_bounds__(256) k_merge_o(const float* __restrict__ opart,
                                                 _Float16* __restrict__ o16) {
  const size_t idx = (size_t)blockIdx.x * 256 + threadIdx.x;  // < BH*N*DH
  constexpr size_t STRIDE = (size_t)BH * N * DH;
  float v = 0.0f;
  #pragma unroll
  for (int s = 0; s < NSL2; ++s) v += opart[s * STRIDE + idx];
  const int bh = (int)(idx / ((size_t)N * DH));
  const size_t rem = idx - (size_t)bh * N * DH;
  const int n = (int)(rem / DH);
  const int d = (int)(rem - (size_t)n * DH);
  const int b = bh >> 2, hd = bh & 3;
  o16[((size_t)b * N + n) * C + hd * DH + d] = (_Float16)v;
}

// ---- output projection: out = O @ Wo + bo  (f32 out)
__global__ void __launch_bounds__(32) k_out(const _Float16* __restrict__ o16,
                                            const _Float16* __restrict__ wot,
                                            const float* __restrict__ bo,
                                            float* __restrict__ out) {
  const int lane = threadIdx.x;
  const int rt = blockIdx.x;     // (B*N)/16
  const int ct = blockIdx.y;     // C/16
  const int lhi = lane >> 4, llo = lane & 15;
  const int co = ct * 16 + llo;
  const _Float16* orow = o16 + (size_t)(rt * 16 + llo) * C;
  const _Float16* wrow = wot + (size_t)co * C;
  v8f acc = {};
  for (int kc = 0; kc < C; kc += 32) {
    AFrag a;
    a.h[0] = *(const v8h*)(orow + kc + (lhi << 4));
    a.h[1] = *(const v8h*)(orow + kc + 8 + (lhi << 4));
    const v16h bw = *(const v16h*)(wrow + kc + (lhi << 4));
    acc = wmma16(a.v, bw, acc);
  }
  const float bias = bo[co];
  #pragma unroll
  for (int r = 0; r < 8; ++r)
    out[(size_t)(rt * 16 + r + (lhi << 3)) * C + co] = acc[r] + bias;
}

extern "C" void kernel_launch(void* const* d_in, const int* in_sizes, int n_in,
                              void* d_out, int out_size, void* d_ws, size_t ws_size,
                              hipStream_t stream) {
  (void)in_sizes; (void)n_in; (void)out_size; (void)ws_size;
  const float* x   = (const float*)d_in[0];
  const float* pe  = (const float*)d_in[1];
  const float* Wq  = (const float*)d_in[2];
  const float* bq  = (const float*)d_in[3];
  const float* Wk  = (const float*)d_in[4];
  const float* bk  = (const float*)d_in[5];
  const float* Wv  = (const float*)d_in[6];
  const float* bv  = (const float*)d_in[7];
  const float* Wo  = (const float*)d_in[8];
  const float* bo  = (const float*)d_in[9];
  const float* lng = (const float*)d_in[10];
  const float* lnb = (const float*)d_in[11];
  float* out = (float*)d_out;
  char* ws = (char*)d_ws;

  _Float16* h16  = (_Float16*)(ws + OFF_H);
  _Float16* q16  = (_Float16*)(ws + OFF_Q);
  _Float16* k16  = (_Float16*)(ws + OFF_K);
  _Float16* vt16 = (_Float16*)(ws + OFF_VT);
  _Float16* o16  = (_Float16*)(ws + OFF_O);
  _Float16* wqt  = (_Float16*)(ws + OFF_WQ);
  _Float16* wkt  = (_Float16*)(ws + OFF_WK);
  _Float16* wvt  = (_Float16*)(ws + OFF_WV);
  _Float16* wot  = (_Float16*)(ws + OFF_WO);
  float* cm  = (float*)(ws + OFF_CM);
  float* cs  = (float*)(ws + OFF_CS);
  float* p1m = (float*)(ws + OFF_P1M);
  float* p1s = (float*)(ws + OFF_P1S);
  float* op  = (float*)(ws + OFF_OP);

  k_trans_hcd<<<(H * C * DH) / 256, 256, 0, stream>>>(Wq, wqt);
  k_trans_hcd<<<(H * C * DH) / 256, 256, 0, stream>>>(Wk, wkt);
  k_trans_hcd<<<(H * C * DH) / 256, 256, 0, stream>>>(Wv, wvt);
  k_trans_cc<<<(C * C) / 256, 256, 0, stream>>>(Wo, wot);
  k_ln<<<B * N, 256, 0, stream>>>(x, pe, lng, lnb, h16);
  k_qkv<<<dim3(N / 16, BH), 32, 0, stream>>>(h16, wqt, wkt, wvt, bq, bk, bv,
                                             q16, k16, vt16);
  k_pass1<<<dim3(N / 32, BH, NSL1), 32, 0, stream>>>(q16, k16, p1m, p1s);
  k_combine<<<(BH * N) / 256, 256, 0, stream>>>(p1m, p1s, cm, cs);
  k_scalev<<<(BH * DH * N) / 256, 256, 0, stream>>>(vt16, cs);
  k_pass2<<<dim3(N / 16, BH, NSL2), 32, 0, stream>>>(q16, k16, vt16, cm, op);
  k_merge_o<<<(BH * N * DH) / 256, 256, 0, stream>>>(op, o16);
  k_out<<<dim3((B * N) / 16, C / 16), 32, 0, stream>>>(o16, wot, bo, out);
}
